// AdvancedActorCritic_85907935855145
// MI455X (gfx1250) — matrix operand order repaired and checked by reading, not compiled, not run on hardware
//
#include <hip/hip_runtime.h>
#include <math.h>

// ---------------------------------------------------------------------------
// Types for CDNA5 WMMA (gfx1250, wave32)
// ---------------------------------------------------------------------------
typedef __attribute__((ext_vector_type(16))) __bf16 v16bf;
typedef __attribute__((ext_vector_type(8)))  __bf16 v8bf;
typedef __attribute__((ext_vector_type(8)))  float  v8f;

#define BATCH_M 16384

// ---- CDNA5 async global->LDS path (guarded; falls back to sync copies) ----
#if defined(__has_builtin)
# if __has_builtin(__builtin_amdgcn_global_load_async_to_lds_b128) && \
     __has_builtin(__builtin_amdgcn_s_wait_asynccnt)
#  define USE_ASYNC_LDS 1
# endif
#endif
#ifndef USE_ASYNC_LDS
# define USE_ASYNC_LDS 0
#endif

#if USE_ASYNC_LDS
typedef int v4i_gcc __attribute__((vector_size(16)));
typedef __attribute__((address_space(1))) v4i_gcc* gptr_v4i;
typedef __attribute__((address_space(3))) v4i_gcc* lptr_v4i;
#endif

__device__ __forceinline__ void copy16B(const __bf16* g, __bf16* l) {
#if USE_ASYNC_LDS
    __builtin_amdgcn_global_load_async_to_lds_b128(
        (gptr_v4i)(void*)g, (lptr_v4i)(void*)l, 0, 0);
#else
    *(v8bf*)l = *(const v8bf*)g;
#endif
}

__device__ __forceinline__ void copy_fence() {
#if USE_ASYNC_LDS
    __builtin_amdgcn_s_wait_asynccnt(0);
#endif
}

__device__ __forceinline__ float gelu_f(float x) {
    float x3 = x * x * x;
    return 0.5f * x * (1.0f + tanhf(0.7978845608028654f * (x + 0.044715f * x3)));
}

// ---------------------------------------------------------------------------
// bf16 WMMA GEMM:  C[M,N] = A[M,K] (bf16) * W[Npad,K] (bf16, zero-padded rows)^T + b
//   MODE 0: v = acc + bias
//   MODE 1: v = aux0 * gelu(acc + bias)                 (gated FFN fuse)
//   MODE 2: s = sigmoid(aux0); v = s*(acc+bias) + (1-s)*aux1 [+ aux2 if HASRES]
// Compile-time epilogue config => straight-line store code, no runtime branches.
// Block: 256 threads = 8 waves; block tile 256(M) x 128(N); K step 32.
// Each wave computes 64x64 = 4x4 WMMA tiles. Double-buffered LDS, async copy.
// ---------------------------------------------------------------------------
#define GBM 256
#define GBN 128
#define GBK 32
#define LDT 40   // bf16 units per LDS row (pad 8 to dodge bank conflicts)

template <int MODE, bool HASF, bool HASB, bool HASRES>
__global__ __launch_bounds__(256) void gemm_wmma_bf16(
    const __bf16* __restrict__ A, const __bf16* __restrict__ W,
    const float* __restrict__ bias,
    int N, int K,
    float* __restrict__ outF, __bf16* __restrict__ outB,
    const float* __restrict__ aux0, const float* __restrict__ aux1,
    const float* __restrict__ aux2)
{
    __shared__ __bf16 sA[2][GBM * LDT];
    __shared__ __bf16 sB[2][GBN * LDT];

    const int tid  = threadIdx.x;
    const int lane = tid & 31;
    const int wave = tid >> 5;

    const int blockM = blockIdx.y * GBM;
    const int blockN = blockIdx.x * GBN;
    const int waveM  = (wave >> 1) * 64;   // 4 waves along M
    const int waveN  = (wave & 1) * 64;    // 2 waves along N

    v8f acc[4][4];
    for (int mi = 0; mi < 4; ++mi)
        for (int ni = 0; ni < 4; ++ni)
            for (int r = 0; r < 8; ++r) acc[mi][ni][r] = 0.0f;

    // cooperative tile copy: A row = tid (4x16B), W row = tid>>1 (2x16B)
    const __bf16* gA = A + (size_t)(blockM + tid) * K;
    const int     brow = tid >> 1;
    const int     boff = (tid & 1) * 16;
    const __bf16* gB = W + (size_t)(blockN + brow) * K + boff;

    const int mr  = lane & 15;
    const int kbA = (lane & 16) ? 8 : 0;   // A frag: e0..7 -> K=kbA.., e8..15 -> K=kbA+16..
    const int khB = (lane & 16) ? 8 : 0;   // B frag: e0..7 -> K=khB.., e8..15 -> K=khB+16..

    const int nsteps = K / GBK;

    // prologue: stage tile 0
    {
        __bf16* la = &sA[0][tid * LDT];
        __bf16* lb = &sB[0][brow * LDT + boff];
        #pragma unroll
        for (int c = 0; c < 4; ++c) copy16B(gA + c * 8, la + c * 8);
        copy16B(gB, lb);
        copy16B(gB + 8, lb + 8);
        copy_fence();
    }
    __syncthreads();

    int buf = 0;
    for (int s = 0; s < nsteps; ++s) {
        const int has_next = (s + 1 < nsteps);
        if (has_next) {
            const int k1 = (s + 1) * GBK;
            __bf16* la = &sA[buf ^ 1][tid * LDT];
            __bf16* lb = &sB[buf ^ 1][brow * LDT + boff];
            #pragma unroll
            for (int c = 0; c < 4; ++c) copy16B(gA + k1 + c * 8, la + c * 8);
            copy16B(gB + k1, lb);
            copy16B(gB + k1 + 8, lb + 8);
        }

        // fragments per ISA 7.12.2 bf16 layouts
        v16bf af[4];
        #pragma unroll
        for (int mi = 0; mi < 4; ++mi) {
            const __bf16* p = &sA[buf][(waveM + mi * 16 + mr) * LDT + kbA];
            v8bf lo = *(const v8bf*)p;
            v8bf hi = *(const v8bf*)(p + 16);
            af[mi] = __builtin_shufflevector(lo, hi, 0,1,2,3,4,5,6,7,8,9,10,11,12,13,14,15);
        }
        #pragma unroll
        for (int ni = 0; ni < 4; ++ni) {
            const __bf16* p = &sB[buf][(waveN + ni * 16 + mr) * LDT + khB];
            v8bf lo = *(const v8bf*)p;
            v8bf hi = *(const v8bf*)(p + 16);
            v16bf bfv = __builtin_shufflevector(lo, hi, 0,1,2,3,4,5,6,7,8,9,10,11,12,13,14,15);
            #pragma unroll
            for (int mi = 0; mi < 4; ++mi)
                acc[mi][ni] = __builtin_amdgcn_wmma_f32_16x16x32_bf16(
                    false, af[mi], false, bfv, (short)0, acc[mi][ni], false, false);
        }

        if (has_next) copy_fence();
        __syncthreads();
        buf ^= 1;
    }

    // epilogue: C layout: col = lane&15, row = (lane>=16 ? 8:0)+r
    const int mrow0 = blockM + waveM + ((lane & 16) ? 8 : 0);
    const int ncol  = lane & 15;
    #pragma unroll
    for (int ni = 0; ni < 4; ++ni) {
        const int n = blockN + waveN + ni * 16 + ncol;
        if (n >= N) continue;                 // uniform-ish column mask (N tails)
        const float bv = bias[n];
        #pragma unroll
        for (int mi = 0; mi < 4; ++mi) {
            #pragma unroll
            for (int r = 0; r < 8; ++r) {
                const int m = mrow0 + mi * 16 + r;
                const size_t idx = (size_t)m * N + n;
                float v = acc[mi][ni][r] + bv;
                if constexpr (MODE == 1) {
                    v = aux0[idx] * gelu_f(v);
                } else if constexpr (MODE == 2) {
                    float sgm = 1.0f / (1.0f + expf(-aux0[idx]));
                    v = sgm * v + (1.0f - sgm) * aux1[idx];
                    if constexpr (HASRES) v += aux2[idx];
                }
                if constexpr (HASF) outF[idx] = v;
                if constexpr (HASB) outB[idx] = (__bf16)v;
            }
        }
    }
}

// ---------------------------------------------------------------------------
// LayerNorm (one wave32 per row):  y = LN(in0 [+ in1]) * g + b; optional gelu.
// ---------------------------------------------------------------------------
__global__ __launch_bounds__(256) void ln_kernel(
    const float* __restrict__ in0, const float* __restrict__ in1,
    const float* __restrict__ gam, const float* __restrict__ bet,
    int D, int do_gelu,
    float* __restrict__ outF, __bf16* __restrict__ outB)
{
    const int wave = threadIdx.x >> 5;
    const int lane = threadIdx.x & 31;
    const int row  = blockIdx.x * 8 + wave;
    const size_t base = (size_t)row * D;
    const int nchunk = D >> 5;   // D/32, <= 16

    float vals[16];
    float sum = 0.0f, ssq = 0.0f;
    for (int j = 0; j < nchunk; ++j) {
        const int c = lane + j * 32;
        float v = in0[base + c];
        if (in1) v += in1[base + c];
        vals[j] = v;
        sum += v;
        ssq += v * v;
    }
    for (int o = 16; o >= 1; o >>= 1) {
        sum += __shfl_xor(sum, o, 32);
        ssq += __shfl_xor(ssq, o, 32);
    }
    const float mean = sum / (float)D;
    const float var  = ssq / (float)D - mean * mean;
    const float rstd = rsqrtf(var + 1e-5f);
    for (int j = 0; j < nchunk; ++j) {
        const int c = lane + j * 32;
        float v = (vals[j] - mean) * rstd * gam[c] + bet[c];
        if (do_gelu) v = gelu_f(v);
        outF[base + c] = v;
        outB[base + c] = (__bf16)v;
    }
}

// f32 -> bf16 bulk convert (activations)
__global__ __launch_bounds__(256) void cvt_bf16_kernel(
    const float* __restrict__ in, __bf16* __restrict__ out, int n)
{
    int i = blockIdx.x * 256 + threadIdx.x;
    if (i < n) out[i] = (__bf16)in[i];
}

// weight f32 -> bf16 with zero padding for rows >= N (row-major, ld = K)
__global__ __launch_bounds__(256) void cvt_w_kernel(
    const float* __restrict__ in, __bf16* __restrict__ out,
    int N, int K, int total)
{
    int i = blockIdx.x * 256 + threadIdx.x;
    if (i >= total) return;
    int row = i / K;
    out[i] = (row < N) ? (__bf16)in[i] : (__bf16)0.0f;
}

// ---------------------------------------------------------------------------
// Host side
// ---------------------------------------------------------------------------
struct LinP { const float* w; const float* b; };
struct LnP  { const float* g; const float* b; };
struct BlockP { LinP wq, wk, wv, wo, ff1, ff2, ff3; LnP ln1, ln2; };
struct HiddenP { LinP layer; LnP norm; LinP gg, gt; };
struct NetP { LinP in_proj; LnP in_norm; BlockP blk[2]; HiddenP hid[2]; LinP out; };
struct NetW { __bf16 *in_proj, *wv[2], *wo[2], *ff1[2], *ff2[2], *ff3[2],
                     *layer[2], *gg[2], *gt[2], *out; };

extern "C" void kernel_launch(void* const* d_in, const int* in_sizes, int n_in,
                              void* d_out, int out_size, void* d_ws, size_t ws_size,
                              hipStream_t stream)
{
    const int M = BATCH_M;
    const float* obs = (const float*)d_in[0];
    (void)in_sizes; (void)n_in; (void)out_size; (void)ws_size;

    // ---- walk params in setup_inputs() dict insertion order ----
    int idx = 1;
    auto takeLin = [&](LinP& l) { l.w = (const float*)d_in[idx++]; l.b = (const float*)d_in[idx++]; };
    auto takeLn  = [&](LnP& l)  { l.g = (const float*)d_in[idx++]; l.b = (const float*)d_in[idx++]; };
    auto takeNet = [&](NetP& n) {
        takeLin(n.in_proj); takeLn(n.in_norm);
        for (int b = 0; b < 2; ++b) {
            takeLin(n.blk[b].wq); takeLin(n.blk[b].wk);
            takeLin(n.blk[b].wv); takeLin(n.blk[b].wo);
            takeLin(n.blk[b].ff1); takeLin(n.blk[b].ff2); takeLin(n.blk[b].ff3);
            takeLn(n.blk[b].ln1); takeLn(n.blk[b].ln2);
        }
        for (int h = 0; h < 2; ++h) {
            takeLin(n.hid[h].layer); takeLn(n.hid[h].norm);
            takeLin(n.hid[h].gg); takeLin(n.hid[h].gt);
        }
        takeLin(n.out);
    };
    NetP actor, critic;
    takeNet(actor);
    takeNet(critic);

    // ---- workspace layout ----
    size_t off = 0;
    auto alloc = [&](size_t bytes) -> void* {
        void* p = (char*)d_ws + off;
        off = (off + bytes + 255) & ~(size_t)255;
        return p;
    };
    __bf16* obs_bf = (__bf16*)alloc((size_t)M * 384 * 2);
    float*  x_f    = (float*) alloc((size_t)M * 512 * 4);
    __bf16* x_b    = (__bf16*)alloc((size_t)M * 512 * 2);
    float*  y_f    = (float*) alloc((size_t)M * 512 * 4);
    __bf16* y_b    = (__bf16*)alloc((size_t)M * 512 * 2);
    float*  t_f    = (float*) alloc((size_t)M * 512 * 4);
    __bf16* v_b    = (__bf16*)alloc((size_t)M * 512 * 2);
    float*  big_f  = (float*) alloc((size_t)M * 2048 * 4);
    __bf16* big_b  = (__bf16*)alloc((size_t)M * 2048 * 2);

    // ---- pre-convert used weights to bf16, rows zero-padded to 128 ----
    auto prepW = [&](const LinP& L, int N, int K) -> __bf16* {
        const int Np = (N + 127) & ~127;
        __bf16* w = (__bf16*)alloc((size_t)Np * K * 2);
        const int total = Np * K;
        cvt_w_kernel<<<(total + 255) / 256, 256, 0, stream>>>(L.w, w, N, K, total);
        return w;
    };
    auto prepNet = [&](const NetP& p, NetW& w, int outN) {
        w.in_proj = prepW(p.in_proj, 512, 384);
        for (int b = 0; b < 2; ++b) {
            w.wv[b]  = prepW(p.blk[b].wv, 512, 512);
            w.wo[b]  = prepW(p.blk[b].wo, 512, 512);
            w.ff1[b] = prepW(p.blk[b].ff1, 2048, 512);
            w.ff2[b] = prepW(p.blk[b].ff2, 2048, 512);
            w.ff3[b] = prepW(p.blk[b].ff3, 512, 2048);
        }
        for (int h = 0; h < 2; ++h) {
            const int din  = 512;
            const int dout = (h == 0) ? 512 : 256;
            w.layer[h] = prepW(p.hid[h].layer, dout, din);
            w.gg[h]    = prepW(p.hid[h].gg, dout, dout);
            w.gt[h]    = prepW(p.hid[h].gt, dout, dout);
        }
        w.out = prepW(p.out, outN, 256);
    };
    NetW actorW, criticW;
    prepNet(actor, actorW, 24);
    prepNet(critic, criticW, 1);

    // ---- GEMM launchers (one template instantiation per epilogue kind) ----
    auto grid_for = [&](int N) { return dim3((N + GBN - 1) / GBN, M / GBM); };

    auto gemm_plainF = [&](const __bf16* A, const __bf16* Wb, const float* b,
                           int N, int K, float* oF) {
        gemm_wmma_bf16<0, true, false, false><<<grid_for(N), 256, 0, stream>>>(
            A, Wb, b, N, K, oF, nullptr, nullptr, nullptr, nullptr);
    };
    auto gemm_plainB = [&](const __bf16* A, const __bf16* Wb, const float* b,
                           int N, int K, __bf16* oB) {
        gemm_wmma_bf16<0, false, true, false><<<grid_for(N), 256, 0, stream>>>(
            A, Wb, b, N, K, nullptr, oB, nullptr, nullptr, nullptr);
    };
    auto gemm_gatedB = [&](const __bf16* A, const __bf16* Wb, const float* b,
                           int N, int K, __bf16* oB, const float* gate) {
        gemm_wmma_bf16<1, false, true, false><<<grid_for(N), 256, 0, stream>>>(
            A, Wb, b, N, K, nullptr, oB, gate, nullptr, nullptr);
    };
    auto gemm_sigres = [&](const __bf16* A, const __bf16* Wb, const float* b,
                           int N, int K, float* oF, __bf16* oB,
                           const float* g, const float* y, const float* res) {
        gemm_wmma_bf16<2, true, true, true><<<grid_for(N), 256, 0, stream>>>(
            A, Wb, b, N, K, oF, oB, g, y, res);
    };
    auto gemm_signores = [&](const __bf16* A, const __bf16* Wb, const float* b,
                             int N, int K, float* oF, __bf16* oB,
                             const float* g, const float* y) {
        gemm_wmma_bf16<2, true, true, false><<<grid_for(N), 256, 0, stream>>>(
            A, Wb, b, N, K, oF, oB, g, y, nullptr);
    };

    auto ln = [&](const float* i0, const float* i1, const LnP& L, int D, int gelu,
                  float* oF, __bf16* oB) {
        ln_kernel<<<M / 8, 256, 0, stream>>>(i0, i1, L.g, L.b, D, gelu, oF, oB);
    };

    // obs -> bf16 once (shared by both networks)
    {
        int n = M * 384;
        cvt_bf16_kernel<<<(n + 255) / 256, 256, 0, stream>>>(obs, obs_bf, n);
    }

    auto run_net = [&](const NetP& net, const NetW& w, float* outp, int outN) {
        // in_proj + in_norm
        gemm_plainF(obs_bf, w.in_proj, net.in_proj.b, 512, 384, t_f);
        ln(t_f, nullptr, net.in_norm, 512, 0, x_f, x_b);

        // transformer blocks (seq len == 1 => attn(x) = wo(wv(x)))
        for (int b = 0; b < 2; ++b) {
            const BlockP& B = net.blk[b];
            gemm_plainB(x_b, w.wv[b], B.wv.b, 512, 512, v_b);
            gemm_plainF(v_b, w.wo[b], B.wo.b, 512, 512, t_f);
            ln(x_f, t_f, B.ln1, 512, 0, x_f, x_b);
            gemm_plainF(x_b, w.ff1[b], B.ff1.b, 2048, 512, big_f);
            gemm_gatedB(x_b, w.ff2[b], B.ff2.b, 2048, 512, big_b, big_f);
            gemm_plainF(big_b, w.ff3[b], B.ff3.b, 512, 2048, t_f);
            ln(x_f, t_f, B.ln2, 512, 0, x_f, x_b);
        }

        // hidden stages with sigmoid gating
        int din = 512;
        for (int h = 0; h < 2; ++h) {
            const HiddenP& H = net.hid[h];
            const int dout = (h == 0) ? 512 : 256;
            gemm_plainF(x_b, w.layer[h], H.layer.b, dout, din, t_f);
            ln(t_f, nullptr, H.norm, dout, 1, y_f, y_b);             // gelu(LN(.))
            gemm_plainF(y_b, w.gg[h], H.gg.b, dout, dout, t_f);
            if (h == 0)
                gemm_sigres(y_b, w.gt[h], H.gt.b, dout, dout, x_f, x_b, t_f, y_f, x_f);
            else
                gemm_signores(y_b, w.gt[h], H.gt.b, dout, dout, x_f, x_b, t_f, y_f);
            din = dout;
        }

        gemm_plainF(x_b, w.out, net.out.b, outN, 256, outp);
    };

    float* action = (float*)d_out;                 // (16384, 24)
    float* value  = action + (size_t)M * 24;       // (16384, 1)
    run_net(actor, actorW, action, 24);
    run_net(critic, criticW, value, 1);
}
